// ModModulatedConv2d_57217554317379
// MI455X (gfx1250) — matrix operand order repaired and checked by reading, not compiled, not run on hardware
//
#include <hip/hip_runtime.h>
#include <hip/hip_bf16.h>
#include <stdint.h>

// Opportunistic CDNA5 async global->LDS path (ASYNCcnt), with safe fallback.
#if defined(__has_builtin)
#  if __has_builtin(__builtin_amdgcn_global_load_async_to_lds_b128) && \
      __has_builtin(__builtin_amdgcn_s_wait_asynccnt)
#    define USE_ASYNC 1
#  endif
#endif
#ifndef USE_ASYNC
#  define USE_ASYNC 0
#endif

typedef __attribute__((ext_vector_type(16))) __bf16 v16bf;
typedef __attribute__((ext_vector_type(8)))  float  v8f;
typedef int v4i __attribute__((vector_size(16)));   // type expected by async builtin

// Problem constants: B=16, Cin=Cout=512, H=W=64, K=3, S=512
#define KB_   16
#define KCIN  512
#define KCOUT 512
#define KHW   4096
#define KSTY  512

#define KSCALE2 (1.0f / 4608.0f)   /* scale^2, scale = 1/sqrt(Cin*3*3) */
#define KSCALE  (1.4731391e-2f)    /* 1/sqrt(4608) */

__device__ __forceinline__ unsigned short f2bf(float f) {
  union { float f; unsigned u; } v; v.f = f;
  unsigned u = v.u;
  u += 0x7FFFu + ((u >> 16) & 1u);                // round-to-nearest-even
  return (unsigned short)(u >> 16);
}

// s[b,ci] = mod_b[ci] + style[b,:] . mod_w[ci,:]
__global__ void k_modstyle(const float* __restrict__ style,
                           const float* __restrict__ mod_w,
                           const float* __restrict__ mod_b,
                           float* __restrict__ s) {
  const int gid = blockIdx.x * blockDim.x + threadIdx.x;   // B*Cin
  const int b = gid >> 9, ci = gid & 511;
  const float* st = style + b * KSTY;
  const float* mw = mod_w + ci * KSTY;
  float acc = mod_b[ci];
  for (int j = 0; j < KSTY; ++j) acc = fmaf(st[j], mw[j], acc);
  s[gid] = acc;
}

// wsq[co,ci] = sum_k weight[co,ci,k]^2
__global__ void k_wsq(const float* __restrict__ weight, float* __restrict__ wsq) {
  const int gid = blockIdx.x * blockDim.x + threadIdx.x;   // Cout*Cin
  const float* wp = weight + gid * 9;
  float a = 0.f;
#pragma unroll
  for (int k = 0; k < 9; ++k) a = fmaf(wp[k], wp[k], a);
  wsq[gid] = a;
}

// demod[b,co] = rsqrt(scale^2 * sum_ci s^2 * wsq + eps)
__global__ void k_demod(const float* __restrict__ s,
                        const float* __restrict__ wsq,
                        float* __restrict__ demod) {
  const int gid = blockIdx.x * blockDim.x + threadIdx.x;   // B*Cout
  const int b = gid >> 9, co = gid & 511;
  const float* sp = s + b * KCIN;
  const float* wq = wsq + co * KCIN;
  float a = 0.f;
  for (int ci = 0; ci < KCIN; ++ci) {
    const float sv = sp[ci];
    a = fmaf(sv * sv, wq[ci], a);
  }
  demod[gid] = rsqrtf(fmaf(a, KSCALE2, 1e-8f));
}

// wT[t][co][ci] = bf16(scale * weight[co][ci][t])
__global__ void k_wt(const float* __restrict__ weight, unsigned short* __restrict__ wT) {
  const int gid = blockIdx.x * blockDim.x + threadIdx.x;   // 9*Cout*Cin
  const int ci = gid & 511, co = (gid >> 9) & 511, t = gid >> 18;
  wT[gid] = f2bf(weight[((co << 9) + ci) * 9 + t] * KSCALE);
}

// xmod NHWC: xm[b][pix][ci] = bf16(x[b][ci][pix] * s[b,ci]); LDS transpose so
// both global reads and writes are coalesced.
__global__ __launch_bounds__(256) void k_xmod(const float* __restrict__ x,
                                              const float* __restrict__ s,
                                              unsigned short* __restrict__ xm) {
  __shared__ unsigned short tile[64][72];                  // [ci][pix], padded
  const int b = blockIdx.z, ct = blockIdx.y, pt = blockIdx.x;   // 16 x 8 x 64
  const int tid = threadIdx.x;
#pragma unroll
  for (int r = 0; r < 16; ++r) {
    const int idx = r * 256 + tid;
    const int ci = idx >> 6, px = idx & 63;
    const int cig = (ct << 6) + ci;
    const float v = x[(((b << 9) + cig) << 12) + (pt << 6) + px] * s[(b << 9) + cig];
    tile[ci][px] = f2bf(v);
  }
  __syncthreads();
#pragma unroll
  for (int r = 0; r < 2; ++r) {
    const int idx = r * 256 + tid;
    const int px = idx >> 3, c8 = (idx & 7) << 3;
    union { unsigned short u[8]; uint4 q; } o;
#pragma unroll
    for (int j = 0; j < 8; ++j) o.u[j] = tile[c8 + j][px];
    *(uint4*)(xm + ((((b << 12) + (pt << 6) + px) << 9) + (ct << 6) + c8)) = o.q;
  }
}

// ---- main implicit-GEMM conv, bf16 WMMA, double-buffered pipeline ----------
// Per batch: M = HW = 4096 (pixels), N = Cout = 512, K = Cin*9 = 4608.
// Block = 256 thr (8 wave32), tile 128x128, wave tile 32x64 (2x4 WMMA accs).
#define BM    128
#define BN    128
#define NSTEP 144      /* 9 taps * 16 channel chunks of 32 */
#define ASTR  48       /* padded bf16 row stride (96B, 16B-aligned) */
#define BSTR  40       /* padded bf16 row stride (80B, 16B-aligned) */

union AFrag { v16bf v; uint4 q[2]; };

__global__ __launch_bounds__(256) void conv_wmma(
    const unsigned short* __restrict__ xmod,  // [B][4096][512] bf16 (s folded)
    const unsigned short* __restrict__ wT,    // [9][Cout][Cin]  bf16 (scale folded)
    const float* __restrict__ demod,          // [B][Cout]
    float* __restrict__ out) {                // [B][Cout][64][64] f32
  __shared__ __align__(16) unsigned short Ab[2][BM * ASTR];
  __shared__ __align__(16) unsigned short Bb[2][BN * BSTR];

  const int mtile = blockIdx.x, ntile = blockIdx.y, b = blockIdx.z;
  const int tid  = threadIdx.x;
  const int lane = tid & 31, wave = tid >> 5;
  const int wm = wave >> 1, wn = wave & 1;        // 4 x 2 wave grid
  const int frow = lane & 15, khalf = lane >> 4;  // fragment lane decomposition

  // A staging: thread -> (pixel m, 16-wide k-half); one 32B contiguous load
  const int am  = tid & 127;
  const int aks = (tid >> 7) << 4;                // 0 or 16
  const int p = mtile * BM + am;
  const int h = p >> 6, w = p & 63;

  // B staging: thread -> (cout row, k-half); one 32B contiguous load
  const int bn  = tid >> 1;
  const int bkh = (tid & 1) << 4;                 // 0 or 16

  // per-step A source pointer + validity (zero-pad at image borders)
  auto a_src = [&](int st, bool& valid) -> const unsigned short* {
    const int t = st >> 4, c0 = (st & 15) << 5;
    const int dh = t / 3 - 1, dw = t - (t / 3) * 3 - 1;
    const int sh = h + dh, sw = w + dw;
    valid = ((unsigned)sh < 64u) & ((unsigned)sw < 64u);
    return xmod + (((b << 12) + (sh << 6) + sw) << 9) + c0 + aks;
  };
  auto b_src = [&](int st) -> const unsigned short* {
    const int t = st >> 4, c0 = (st & 15) << 5;
    return wT + ((((t << 9) + ntile * BN + bn) << 9) + c0 + bkh);
  };

  const uint4 z4 = {0u, 0u, 0u, 0u};
  uint4 ar0 = z4, ar1 = z4;
#if !USE_ASYNC
  uint4 br0 = z4, br1 = z4;
#endif

  auto load_next = [&](int st) {
    bool valid;
    const unsigned short* ap = a_src(st, valid);
    ar0 = z4; ar1 = z4;
    if (valid) { const uint4* g = (const uint4*)ap; ar0 = g[0]; ar1 = g[1]; }
#if USE_ASYNC
    const unsigned short* bp = b_src(st);
    unsigned short* lp = &Bb[(st & 1)][bn * BSTR + bkh];
    __builtin_amdgcn_global_load_async_to_lds_b128((v4i*)bp,       (v4i*)lp,       0, 0);
    __builtin_amdgcn_global_load_async_to_lds_b128((v4i*)(bp + 8), (v4i*)(lp + 8), 0, 0);
#else
    const uint4* g = (const uint4*)b_src(st);
    br0 = g[0]; br1 = g[1];
#endif
  };
  auto commit_next = [&](int st) {
    uint4* ad = (uint4*)&Ab[(st & 1)][am * ASTR + aks];
    ad[0] = ar0; ad[1] = ar1;
#if !USE_ASYNC
    uint4* bd = (uint4*)&Bb[(st & 1)][bn * BSTR + bkh];
    bd[0] = br0; bd[1] = br1;
#endif
  };

  v8f acc[2][4];
  const v8f vzero = {0.f, 0.f, 0.f, 0.f, 0.f, 0.f, 0.f, 0.f};
#pragma unroll
  for (int i = 0; i < 2; ++i)
#pragma unroll
    for (int j = 0; j < 4; ++j) acc[i][j] = vzero;

  // prologue: stage step 0
  load_next(0);
  commit_next(0);
#if USE_ASYNC
  __builtin_amdgcn_s_wait_asynccnt(0);
#endif
  __syncthreads();

#pragma unroll 2
  for (int st = 0; st < NSTEP; ++st) {
    const int cb = st & 1;
    const bool more = (st + 1) < NSTEP;
    if (more) load_next(st + 1);      // global loads overlap the WMMAs below

    // Fragment loads per ISA 16-bit layouts:
    // A 16x32: lanes 0-15 hold K{0..7,16..23}, lanes 16-31 K{8..15,24..31}
    // B 32x16: lanes 0-15 col n K{0..7,16..23}, lanes 16-31 col n K{8..15,24..31}
    AFrag a[2], bfr[4];
#pragma unroll
    for (int tm = 0; tm < 2; ++tm) {
      const unsigned short* ap = &Ab[cb][(wm * 32 + tm * 16 + frow) * ASTR + khalf * 8];
      a[tm].q[0] = *(const uint4*)ap;
      a[tm].q[1] = *(const uint4*)(ap + 16);
    }
#pragma unroll
    for (int tn = 0; tn < 4; ++tn) {
      const unsigned short* bp = &Bb[cb][(wn * 64 + tn * 16 + frow) * BSTR + khalf * 8];
      bfr[tn].q[0] = *(const uint4*)bp;
      bfr[tn].q[1] = *(const uint4*)(bp + 16);
    }
#pragma unroll
    for (int tm = 0; tm < 2; ++tm)
#pragma unroll
      for (int tn = 0; tn < 4; ++tn)
        acc[tm][tn] = __builtin_amdgcn_wmma_f32_16x16x32_bf16(
            false, a[tm].v, false, bfr[tn].v, (short)0, acc[tm][tn], false, false);

    if (more) commit_next(st + 1);    // write other buffer (safe: barrier below)
#if USE_ASYNC
    __builtin_amdgcn_s_wait_asynccnt(0);
#endif
    __syncthreads();
  }

  // epilogue: C/D layout -> lanes 0-15: M=r,N=lane; lanes 16-31: M=8+r,N=lane-16
#pragma unroll
  for (int tn = 0; tn < 4; ++tn) {
    const int co = ntile * BN + wn * 64 + tn * 16 + frow;
    const float dm = demod[(b << 9) + co];
#pragma unroll
    for (int tm = 0; tm < 2; ++tm) {
      const int pbase = mtile * BM + wm * 32 + tm * 16 + khalf * 8;
      float* op = out + (((b << 9) + co) << 12) + pbase;
#pragma unroll
      for (int r = 0; r < 8; ++r)
        op[r] = acc[tm][tn][r] * dm;
    }
  }
}

extern "C" void kernel_launch(void* const* d_in, const int* in_sizes, int n_in,
                              void* d_out, int out_size, void* d_ws, size_t ws_size,
                              hipStream_t stream) {
  (void)in_sizes; (void)n_in; (void)out_size; (void)ws_size;
  const float* input  = (const float*)d_in[0];
  const float* style  = (const float*)d_in[1];
  const float* weight = (const float*)d_in[2];
  const float* mod_w  = (const float*)d_in[3];
  const float* mod_b  = (const float*)d_in[4];
  float* out = (float*)d_out;

  // workspace: s 32K | demod 32K | wsq 1M | wT 4.5M | xmod 64M  (~73 MB)
  char* ws = (char*)d_ws;
  float*          s     = (float*)(ws);
  float*          demod = (float*)(ws + (32u << 10));
  float*          wsq   = (float*)(ws + (64u << 10));
  unsigned short* wT    = (unsigned short*)(ws + (64u << 10) + (1u << 20));
  unsigned short* xmod  = (unsigned short*)(ws + (64u << 10) + (1u << 20) + 4718592u);

  k_modstyle<<<KB_ * KCIN / 256,       256, 0, stream>>>(style, mod_w, mod_b, s);
  k_wsq     <<<KCOUT * KCIN / 256,     256, 0, stream>>>(weight, wsq);
  k_demod   <<<KB_ * KCOUT / 256,      256, 0, stream>>>(s, wsq, demod);
  k_wt      <<<9 * KCOUT * KCIN / 256, 256, 0, stream>>>(weight, wT);
  { dim3 g(64, 8, KB_); k_xmod<<<g, 256, 0, stream>>>(input, s, xmod); }

  dim3 grid(KHW / BM, KCOUT / BN, KB_);   // 32 x 4 x 16 = 2048 blocks
  conv_wmma<<<grid, 256, 0, stream>>>(xmod, wT, demod, out);
}
